// SelectiveSSM_56959856280343
// MI455X (gfx1250) — matrix operand order repaired and checked by reading, not compiled, not run on hardware
//
#include <hip/hip_runtime.h>
#include <hip/hip_bf16.h>
#include <math.h>
#include <stdint.h>

// ---------------------------------------------------------------------------
// SelectiveSSM for MI455X (gfx1250), wave32 + WMMA + async-to-LDS.
//  Phase 0: one-shot f32 -> split-bf16 (hi/lo) conversion of X and weights.
//  Phase 1: three split-bf16 (3x v_wmma_f32_16x16x32_bf16) GEMMs with
//           double-buffered GLOBAL_LOAD_ASYNC_TO_LDS_B128 staging:
//     delta_eff = softplus(x@W_delta^T + b_delta + delta_bias)   (4096 x   768)
//     B_tot     = x@W_B^T + b_B + Bmat                            (4096 x 12288)
//     C_tot     = x@W_C^T + b_C + C                               (4096 x 12288)
//  Phase 2: recurrent scan, thread=(b,d,n), shfl reduction over n.
// Workspace: ~505MB in d_ws.
// ---------------------------------------------------------------------------

typedef __attribute__((ext_vector_type(16))) __bf16         v16bf;
typedef __attribute__((ext_vector_type(8)))  float          v8f;
typedef __attribute__((ext_vector_type(8)))  unsigned short u16x8;

union FragBF { v16bf v; u16x8 h[2]; };

__device__ __forceinline__ unsigned short f32_to_bf16_rne(float f) {
  unsigned int u = __float_as_uint(f);
  u += 0x7FFFu + ((u >> 16) & 1u);
  return (unsigned short)(u >> 16);
}
__device__ __forceinline__ float bf16_bits_to_f32(unsigned short h) {
  return __uint_as_float(((unsigned int)h) << 16);
}
__device__ __forceinline__ void cvt_split(float f, unsigned short& hi, unsigned short& lo) {
  hi = f32_to_bf16_rne(f);
  lo = f32_to_bf16_rne(f - bf16_bits_to_f32(hi));
}

// 16B global -> LDS async copy (GV mode, per-lane addresses, ASYNCcnt).
__device__ __forceinline__ void async_copy16(unsigned lds_off, const void* gsrc) {
  unsigned long long ga = (unsigned long long)(uintptr_t)gsrc;
  asm volatile("global_load_async_to_lds_b128 %0, %1, off"
               :: "v"(lds_off), "v"(ga) : "memory");
}
__device__ __forceinline__ unsigned lds_off_of(const void* p) {
  return (unsigned)(uintptr_t)p;   // low 32 bits of generic ptr = LDS offset
}

// ---------------------------------------------------------------------------
// Phase 0: elementwise split conversion (vectorized x4).
// ---------------------------------------------------------------------------
__global__ __launch_bounds__(256)
void convert_split(const float* __restrict__ src, unsigned short* __restrict__ hi,
                   unsigned short* __restrict__ lo, int n4)
{
  int i = blockIdx.x * 256 + threadIdx.x;
  if (i >= n4) return;
  float4 v = ((const float4*)src)[i];
  ushort4 h4, l4;
  cvt_split(v.x, h4.x, l4.x);
  cvt_split(v.y, h4.y, l4.y);
  cvt_split(v.z, h4.z, l4.z);
  cvt_split(v.w, h4.w, l4.w);
  ((ushort4*)hi)[i] = h4;
  ((ushort4*)lo)[i] = l4;
}

// ---------------------------------------------------------------------------
// GEMM: out[m, col] = sum_k X[m,k] * W[col,k]  (+ bias[col] + extra[col])
//       split-bf16, double-buffered async-to-LDS staging.
// ---------------------------------------------------------------------------
#define TM   64
#define TN   128
#define TK   32
#define LDA  40   // shorts per LDS row (32 + 8 pad); rows stay 16B-aligned

// issue 6 async b128 copies per thread: A-hi, A-lo (1 each), B-hi, B-lo (2 each)
__device__ __forceinline__ void issue_stage(
    const unsigned short* __restrict__ Xhi, const unsigned short* __restrict__ Xlo,
    const unsigned short* __restrict__ Whi, const unsigned short* __restrict__ Wlo,
    unsigned short* sAh, unsigned short* sAl,
    unsigned short* sBh, unsigned short* sBl,
    int tid, int mbase, int nbase, int K, int k0)
{
  {
    int r = tid >> 2, c = (tid & 3) * 8;            // 64 rows x 4 chunks
    size_t s = (size_t)(mbase + r) * K + k0 + c;
    unsigned dh = lds_off_of(&sAh[r * LDA + c]);
    unsigned dl = lds_off_of(&sAl[r * LDA + c]);
    async_copy16(dh, &Xhi[s]);
    async_copy16(dl, &Xlo[s]);
  }
#pragma unroll
  for (int it = 0; it < 2; ++it) {                  // 128 rows x 4 chunks
    int q = tid + it * 256;
    int r = q >> 2, c = (q & 3) * 8;
    size_t s = (size_t)(nbase + r) * K + k0 + c;
    unsigned dh = lds_off_of(&sBh[r * LDA + c]);
    unsigned dl = lds_off_of(&sBl[r * LDA + c]);
    async_copy16(dh, &Whi[s]);
    async_copy16(dl, &Wlo[s]);
  }
}

template<bool SOFTPLUS>
__global__ __launch_bounds__(256)
void gemm_bf16_async(const unsigned short* __restrict__ Xhi, const unsigned short* __restrict__ Xlo,
                     const unsigned short* __restrict__ Whi, const unsigned short* __restrict__ Wlo,
                     const float* __restrict__ bias, const float* __restrict__ extra,
                     float* __restrict__ out, int K, int Ncols)
{
  __shared__ unsigned short sAhi[2][TM * LDA], sAlo[2][TM * LDA];
  __shared__ unsigned short sBhi[2][TN * LDA], sBlo[2][TN * LDA];

  const int tid   = threadIdx.x;
  const int mbase = blockIdx.y * TM;
  const int nbase = blockIdx.x * TN;

  const int wid  = tid >> 5;
  const int lane = tid & 31;
  const int wm   = wid & 1;        // 2 waves along M  -> 32 rows each
  const int wn   = wid >> 1;       // 4 waves along N  -> 32 cols each
  const int lr   = lane & 15;
  const int hb   = lane >> 4;      // lane half

  v8f acc[2][2];
#pragma unroll
  for (int i = 0; i < 2; ++i)
#pragma unroll
    for (int j = 0; j < 2; ++j)
#pragma unroll
      for (int e = 0; e < 8; ++e) acc[i][j][e] = 0.0f;

  const int nk = K / TK;
  issue_stage(Xhi, Xlo, Whi, Wlo, sAhi[0], sAlo[0], sBhi[0], sBlo[0],
              tid, mbase, nbase, K, 0);

  for (int kt = 0; kt < nk; ++kt) {
    const int cur = kt & 1;
    if (kt + 1 < nk) {
      issue_stage(Xhi, Xlo, Whi, Wlo, sAhi[cur ^ 1], sAlo[cur ^ 1],
                  sBhi[cur ^ 1], sBlo[cur ^ 1], tid, mbase, nbase, K, (kt + 1) * TK);
      asm volatile("s_wait_asynccnt 0x6" ::: "memory");  // retire current buffer
    } else {
      asm volatile("s_wait_asynccnt 0x0" ::: "memory");
    }
    __syncthreads();

    const unsigned short* pAh = sAhi[cur];
    const unsigned short* pAl = sAlo[cur];
    const unsigned short* pBh = sBhi[cur];
    const unsigned short* pBl = sBlo[cur];

    // ---- fragment loads (ISA 16-bit A 16x32 / B 32x16 layouts) ----
    FragBF ahi[2], alo[2], bhi[2], blo[2];
#pragma unroll
    for (int i = 0; i < 2; ++i) {
      int row = wm * 32 + i * 16 + lr;
      ahi[i].h[0] = *(const u16x8*)&pAh[row * LDA + 8 * hb];
      ahi[i].h[1] = *(const u16x8*)&pAh[row * LDA + 16 + 8 * hb];
      alo[i].h[0] = *(const u16x8*)&pAl[row * LDA + 8 * hb];
      alo[i].h[1] = *(const u16x8*)&pAl[row * LDA + 16 + 8 * hb];
    }
#pragma unroll
    for (int j = 0; j < 2; ++j) {
      int nn = wn * 32 + j * 16 + lr;
      bhi[j].h[0] = *(const u16x8*)&pBh[nn * LDA + 8 * hb];
      bhi[j].h[1] = *(const u16x8*)&pBh[nn * LDA + 16 + 8 * hb];
      blo[j].h[0] = *(const u16x8*)&pBl[nn * LDA + 8 * hb];
      blo[j].h[1] = *(const u16x8*)&pBl[nn * LDA + 16 + 8 * hb];
    }

    // ---- 12 WMMAs: small terms first, then hi*hi ----
#pragma unroll
    for (int i = 0; i < 2; ++i)
#pragma unroll
      for (int j = 0; j < 2; ++j) {
        acc[i][j] = __builtin_amdgcn_wmma_f32_16x16x32_bf16(
            false, alo[i].v, false, bhi[j].v, (short)0, acc[i][j], false, false);
        acc[i][j] = __builtin_amdgcn_wmma_f32_16x16x32_bf16(
            false, ahi[i].v, false, blo[j].v, (short)0, acc[i][j], false, false);
        acc[i][j] = __builtin_amdgcn_wmma_f32_16x16x32_bf16(
            false, ahi[i].v, false, bhi[j].v, (short)0, acc[i][j], false, false);
      }
    __syncthreads();   // protect buffer 'cur' before it is re-staged
  }

  // ---- epilogue: + bias + extra (+softplus), C/D layout m = e + 8*hb ----
#pragma unroll
  for (int i = 0; i < 2; ++i)
#pragma unroll
    for (int j = 0; j < 2; ++j) {
      int col   = nbase + wn * 32 + j * 16 + lr;
      float add = bias[col] + extra[col];
#pragma unroll
      for (int e = 0; e < 8; ++e) {
        int row = mbase + wm * 32 + i * 16 + e + 8 * hb;
        float v = acc[i][j][e] + add;
        if (SOFTPLUS) {
          v = (v > 20.0f) ? v : log1pf(__expf(v));
          out[(size_t)row * Ncols + col] = v;          // deff: small, L2-resident
        } else {
          __builtin_nontemporal_store(v, &out[(size_t)row * Ncols + col]); // streamed
        }
      }
    }
}

// ---------------------------------------------------------------------------
// Scan: h[l] = a*h[l-1] + bx;  y = sum_n c*h.  thread=(b,d,n), 16-lane shfl sum.
// ---------------------------------------------------------------------------
#define SSM_L 2048
#define SSM_D 768
#define SSM_DN 12288
#define SCAN_T 8

__device__ __forceinline__ void scan_load(const float* __restrict__ Bt,
                                          const float* __restrict__ Ct,
                                          const float* __restrict__ De,
                                          const float* __restrict__ X,
                                          size_t rowBase, int l0, int dn, int d,
                                          float (&bt)[SCAN_T], float (&ct)[SCAN_T],
                                          float (&de)[SCAN_T], float (&xv)[SCAN_T])
{
#pragma unroll
  for (int t = 0; t < SCAN_T; ++t) {
    size_t r = rowBase + (size_t)(l0 + t);
    bt[t] = __builtin_nontemporal_load(&Bt[r * SSM_DN + dn]);  // read-once streams
    ct[t] = __builtin_nontemporal_load(&Ct[r * SSM_DN + dn]);
    de[t] = De[r * SSM_D + d];
    xv[t] = X[r * SSM_D + d];
  }
}

__global__ __launch_bounds__(256)
void ssm_scan(const float* __restrict__ De, const float* __restrict__ Bt,
              const float* __restrict__ Ct, const float* __restrict__ X,
              const float* __restrict__ A, float* __restrict__ Y)
{
  const int tid = threadIdx.x;
  const int n   = tid & 15;
  const int dl  = tid >> 4;
  const int b   = blockIdx.x / 48;
  const int d   = (blockIdx.x % 48) * 16 + dl;
  const int dn  = d * 16 + n;

  const float Av  = A[dn];
  const float inv = 1.0f / (Av + 1e-6f);
  const size_t rowBase = (size_t)b * SSM_L;

  float h = 0.0f;
  float bt0[SCAN_T], ct0[SCAN_T], de0[SCAN_T], xv0[SCAN_T];
  float bt1[SCAN_T], ct1[SCAN_T], de1[SCAN_T], xv1[SCAN_T];

  auto compute = [&](int l0, const float (&bt)[SCAN_T], const float (&ct)[SCAN_T],
                     const float (&de)[SCAN_T], const float (&xv)[SCAN_T]) {
#pragma unroll
    for (int t = 0; t < SCAN_T; ++t) {
      float a  = __expf(de[t] * Av);                 // A_bar
      float bx = (a - 1.0f) * inv * bt[t] * xv[t];   // B_bar * x
      h = fmaf(a, h, bx);
      float p = ct[t] * h;
      p += __shfl_xor(p, 1, 32);
      p += __shfl_xor(p, 2, 32);
      p += __shfl_xor(p, 4, 32);
      p += __shfl_xor(p, 8, 32);                     // sum over n within 16 lanes
      if (n == 0) Y[(rowBase + (size_t)(l0 + t)) * SSM_D + d] = p;
    }
  };

  scan_load(Bt, Ct, De, X, rowBase, 0, dn, d, bt0, ct0, de0, xv0);
  for (int l0 = 0; l0 < SSM_L; l0 += 2 * SCAN_T) {
    scan_load(Bt, Ct, De, X, rowBase, l0 + SCAN_T, dn, d, bt1, ct1, de1, xv1);
    compute(l0, bt0, ct0, de0, xv0);
    if (l0 + 2 * SCAN_T < SSM_L)
      scan_load(Bt, Ct, De, X, rowBase, l0 + 2 * SCAN_T, dn, d, bt0, ct0, de0, xv0);
    compute(l0 + SCAN_T, bt1, ct1, de1, xv1);
  }
}

// ---------------------------------------------------------------------------
extern "C" void kernel_launch(void* const* d_in, const int* in_sizes, int n_in,
                              void* d_out, int out_size, void* d_ws, size_t ws_size,
                              hipStream_t stream) {
  const float* x          = (const float*)d_in[0];   // (2,2048,768)
  const float* A          = (const float*)d_in[1];   // (768,16)
  const float* Bmat       = (const float*)d_in[2];   // (768,16)
  const float* Cmat       = (const float*)d_in[3];   // (768,16)
  const float* delta_bias = (const float*)d_in[4];   // (768,)
  const float* W_delta    = (const float*)d_in[5];   // (768,768)
  const float* b_delta    = (const float*)d_in[6];   // (768,)
  const float* W_B        = (const float*)d_in[7];   // (12288,768)
  const float* b_B        = (const float*)d_in[8];   // (12288,)
  const float* W_C        = (const float*)d_in[9];   // (12288,768)
  const float* b_C        = (const float*)d_in[10];  // (12288,)
  float* y = (float*)d_out;                          // (2,2048,768)

  const int M = 2 * SSM_L;       // 4096
  const int K = SSM_D;           // 768

  // ---- workspace carve-up ----
  char* ws = (char*)d_ws;
  size_t off = 0;
  auto carve = [&](size_t bytes) { void* p = ws + off; off += (bytes + 255) & ~(size_t)255; return p; };
  float*          deff  = (float*)         carve((size_t)M * SSM_D  * 4);
  float*          Btot  = (float*)         carve((size_t)M * SSM_DN * 4);
  float*          Ctot  = (float*)         carve((size_t)M * SSM_DN * 4);
  unsigned short* Xhi   = (unsigned short*)carve((size_t)M * SSM_D  * 2);
  unsigned short* Xlo   = (unsigned short*)carve((size_t)M * SSM_D  * 2);
  unsigned short* Wd_hi = (unsigned short*)carve((size_t)SSM_D * SSM_D * 2);
  unsigned short* Wd_lo = (unsigned short*)carve((size_t)SSM_D * SSM_D * 2);
  unsigned short* WB_hi = (unsigned short*)carve((size_t)SSM_DN * SSM_D * 2);
  unsigned short* WB_lo = (unsigned short*)carve((size_t)SSM_DN * SSM_D * 2);
  unsigned short* WC_hi = (unsigned short*)carve((size_t)SSM_DN * SSM_D * 2);
  unsigned short* WC_lo = (unsigned short*)carve((size_t)SSM_DN * SSM_D * 2);

  dim3 blk(256);
  // ---- phase 0: split conversions ----
  {
    int n4;
    n4 = (M * SSM_D) / 4;
    convert_split<<<dim3((n4 + 255) / 256), blk, 0, stream>>>(x, Xhi, Xlo, n4);
    n4 = (SSM_D * SSM_D) / 4;
    convert_split<<<dim3((n4 + 255) / 256), blk, 0, stream>>>(W_delta, Wd_hi, Wd_lo, n4);
    n4 = (SSM_DN * SSM_D) / 4;
    convert_split<<<dim3((n4 + 255) / 256), blk, 0, stream>>>(W_B, WB_hi, WB_lo, n4);
    convert_split<<<dim3((n4 + 255) / 256), blk, 0, stream>>>(W_C, WC_hi, WC_lo, n4);
  }

  // ---- phase 1: WMMA GEMMs ----
  gemm_bf16_async<true><<<dim3(SSM_D / TN, M / TM), blk, 0, stream>>>(
      Xhi, Xlo, Wd_hi, Wd_lo, b_delta, delta_bias, deff, K, SSM_D);
  gemm_bf16_async<false><<<dim3(SSM_DN / TN, M / TM), blk, 0, stream>>>(
      Xhi, Xlo, WB_hi, WB_lo, b_B, Bmat, Btot, K, SSM_DN);
  gemm_bf16_async<false><<<dim3(SSM_DN / TN, M / TM), blk, 0, stream>>>(
      Xhi, Xlo, WC_hi, WC_lo, b_C, Cmat, Ctot, K, SSM_DN);

  // ---- phase 2: recurrence ----
  ssm_scan<<<dim3(96), blk, 0, stream>>>(deff, Btot, Ctot, x, A, y);
}
